// EdgeNet_4964982194217
// MI455X (gfx1250) — matrix operand order repaired and checked by reading, not compiled, not run on hardware
//
#include <hip/hip_runtime.h>
#include <hip/hip_bf16.h>

#define N_NODES  8192
#define N_EDGES  24576
#define D_FEAT   64
#define N_QUBITS 16
#define PI_F     3.14159265358979f

typedef __attribute__((ext_vector_type(16))) _Float16 v16h;
typedef __attribute__((ext_vector_type(8)))  float    v8f;
typedef __attribute__((ext_vector_type(4)))  float    f4;

// ---------------------------------------------------------------------------
// Kernel 0: initialize global min/max accumulators (ws is poisoned with 0xAA).
// h >= 0 after ReLU, so non-negative floats order correctly as uints.
// ---------------------------------------------------------------------------
__global__ void EdgeNet_init_minmax(unsigned int* mm) {
    if (threadIdx.x == 0) {
        mm[0] = 0x7F800000u;  // +inf  (min accumulator)
        mm[1] = 0x00000000u;  // 0.0f  (max accumulator)
    }
}

// ---------------------------------------------------------------------------
// Kernel 1: recover per-edge node index from a one-hot incidence matrix.
// Streams 805 MB with non-temporal float4 loads (this is the roofline cost).
// Exactly one nonzero per column -> unordered scattered stores are race-free.
// ---------------------------------------------------------------------------
__global__ void EdgeNet_extract_idx(const float* __restrict__ M,
                                    int* __restrict__ idx) {
    const long long nvec   = (long long)N_NODES * (long long)N_EDGES / 4;
    const long long stride = (long long)gridDim.x * blockDim.x;
    for (long long t = (long long)blockIdx.x * blockDim.x + threadIdx.x;
         t < nvec; t += stride) {
        f4 v = __builtin_nontemporal_load((const f4*)M + t);
        if (v.x != 0.f || v.y != 0.f || v.z != 0.f || v.w != 0.f) {
            long long base = t * 4;                 // N_EDGES % 4 == 0 -> one row
            int row = (int)(base / N_EDGES);
            int col = (int)(base - (long long)row * N_EDGES);
            if (v.x != 0.f) idx[col]     = row;
            if (v.y != 0.f) idx[col + 1] = row;
            if (v.z != 0.f) idx[col + 2] = row;
            if (v.w != 0.f) idx[col + 3] = row;
        }
    }
}

// ---------------------------------------------------------------------------
// Kernel 2: per 16-edge tile (one wave32 each):
//   B[16x128] = concat(X[idx_o], X[idx_i])  (gather, f32 -> f16)
//   h = relu(B @ W_in + b_in)  via 4x v_wmma_f32_16x16x32_f16 (f32 accum)
// then wave min/max reduction + global atomic combine.
// Fragment layouts follow CDNA5 ISA 7.12.2 (16-bit A 16x32, B 32x16, f32 C/D).
// ---------------------------------------------------------------------------
__global__ void EdgeNet_mlp_wmma(const float* __restrict__ X,
                                 const int*   __restrict__ idx_o,   // from Ro
                                 const int*   __restrict__ idx_i,   // from Ri
                                 const float* __restrict__ W,       // [128,16]
                                 const float* __restrict__ bvec,    // [16]
                                 float*       __restrict__ out,     // [E,16]
                                 unsigned int* __restrict__ mm) {
    const int lane = threadIdx.x & 31;
    const int wave = threadIdx.x >> 5;
    const int tile = blockIdx.x * (blockDim.x >> 5) + wave;
    const int eb   = tile * 16;
    const int m    = lane & 15;            // M row (A) / N col (B,C,D)
    const int hi   = lane >> 4;            // lane half selector
    const int e    = eb + m;

    const float* __restrict__ Xo = X + (long long)idx_o[e] * D_FEAT;
    const float* __restrict__ Xi = X + (long long)idx_i[e] * D_FEAT;

    v8f c = {};
#pragma unroll
    for (int s = 0; s < 4; ++s) {
        union { _Float16 h[16]; v16h v; } A, B;
        // A fragment: 16-bit A 16x32 layout.
        //   lanes 0-15: K in {0..7, 16..23}; lanes 16-31: K in {8..15, 24..31}
#pragma unroll
        for (int j = 0; j < 16; ++j) {
            int vg = j >> 1;
            int kk = ((vg >= 4) ? 16 : 0) + (hi ? 8 : 0) + ((vg & 3) * 2) + (j & 1);
            int k  = 32 * s + kk;          // 0..127 over the 4 steps
            float f = (k < D_FEAT) ? Xo[k] : Xi[k - D_FEAT];
            A.h[j] = (_Float16)f;
        }
        // B fragment: 16-bit B 32x16 layout.
        //   lanes 0-15: K in {0..7, 16..23}; lanes 16-31: K in {8..15, 24..31}, column n = m
#pragma unroll
        for (int j = 0; j < 16; ++j) {
            int k = 32 * s + ((j >= 8) ? 16 : 0) + (hi ? 8 : 0) + (j & 7);
            B.h[j] = (_Float16)W[k * N_QUBITS + m];
        }
        c = __builtin_amdgcn_wmma_f32_16x16x32_f16(
                /*neg_a=*/false, A.v, /*neg_b=*/false, B.v,
                /*c_mod=*/(short)0, c, /*reuse_a=*/false, /*reuse_b=*/false);
    }

    // C/D layout: VGPR r -> row M = r + 8*hi, column N = m.
    const float bias = bvec[m];
    float lmin = __builtin_inff();
    float lmax = 0.f;
#pragma unroll
    for (int r = 0; r < 8; ++r) {
        int   Mrow = r + hi * 8;
        float v    = fmaxf(c[r] + bias, 0.f);
        out[(eb + Mrow) * N_QUBITS + m] = v;
        lmin = fminf(lmin, v);
        lmax = fmaxf(lmax, v);
    }
    // wave32 butterfly reduction
#pragma unroll
    for (int off = 16; off > 0; off >>= 1) {
        lmin = fminf(lmin, __shfl_xor(lmin, off, 32));
        lmax = fmaxf(lmax, __shfl_xor(lmax, off, 32));
    }
    if (lane == 0) {
        atomicMin(&mm[0], __float_as_uint(lmin));
        atomicMax(&mm[1], __float_as_uint(lmax));
    }
}

// ---------------------------------------------------------------------------
// Kernel 3: global min-max rescale to [0, pi], in place on d_out.
// ---------------------------------------------------------------------------
__global__ void EdgeNet_normalize(float* __restrict__ out,
                                  const unsigned int* __restrict__ mm, int n) {
    const float lo    = __uint_as_float(mm[0]);
    const float hicap = __uint_as_float(mm[1]);
    const float scale = PI_F / (hicap - lo);
    int i = blockIdx.x * blockDim.x + threadIdx.x;
    if (i < n) out[i] = (out[i] - lo) * scale;
}

extern "C" void kernel_launch(void* const* d_in, const int* in_sizes, int n_in,
                              void* d_out, int out_size, void* d_ws, size_t ws_size,
                              hipStream_t stream) {
    const float* X   = (const float*)d_in[0];   // [8192, 64]
    const float* Ri  = (const float*)d_in[1];   // [8192, 24576]
    const float* Ro  = (const float*)d_in[2];   // [8192, 24576]
    const float* Win = (const float*)d_in[3];   // [128, 16]
    const float* bin = (const float*)d_in[4];   // [16]
    float* out = (float*)d_out;                 // [24576, 16]

    int* idx_o = (int*)d_ws;                    // bo = Ro.T @ X  -> first half
    int* idx_i = idx_o + N_EDGES;               // bi = Ri.T @ X  -> second half
    unsigned int* mm = (unsigned int*)(idx_i + N_EDGES);

    EdgeNet_init_minmax<<<1, 32, 0, stream>>>(mm);
    EdgeNet_extract_idx<<<8192, 256, 0, stream>>>(Ro, idx_o);
    EdgeNet_extract_idx<<<8192, 256, 0, stream>>>(Ri, idx_i);
    // 24576 edges / 16 per wave = 1536 waves; 4 waves (128 thr) per block.
    EdgeNet_mlp_wmma<<<N_EDGES / 16 / 4, 128, 0, stream>>>(
        X, idx_o, idx_i, Win, bin, out, mm);
    EdgeNet_normalize<<<(N_EDGES * N_QUBITS + 255) / 256, 256, 0, stream>>>(
        out, mm, N_EDGES * N_QUBITS);
}
